// OrderSensitiveMetricLoss_60069412602555
// MI455X (gfx1250) — compile-verified, matrix operand order checked by reading, not run on hardware
//
#include <hip/hip_runtime.h>
#include <hip/hip_bf16.h>

typedef __attribute__((ext_vector_type(16))) _Float16 v16h;
typedef __attribute__((ext_vector_type(8)))  float    v8f;

#define NN    256
#define NHASH 64
#define NCLS  14

// Load one A/B fragment for v_wmma_f32_16x16x32_f16, per CDNA5 ISA VGPR layout:
// lane l holds matrix row (l&15); element e of the v16h maps to
// K = kbase + (e&7) + 8*hi + 16*(e>>3), hi = l>>4 (provably 0/1).
// FULL=true: K provably < KMAX -> unconditional contiguous loads.
// FULL=false: clamp index in-bounds, then branchless select of 0 (no exec branches).
template<int KMAX, bool FULL>
__device__ __forceinline__ v16h frag_load_row(const float* __restrict__ row,
                                              int hi, int kbase) {
  v16h f;
#pragma unroll
  for (int e = 0; e < 16; ++e) {
    int k = kbase + (e & 7) + 8 * hi + 16 * (e >> 3);
    float v;
    if (FULL) {
      v = row[k];                        // k <= kbase+31 < KMAX, compiler can prove
    } else {
      int kc = k < (KMAX - 1) ? k : (KMAX - 1);   // always-legal address
      v = row[kc];
      v = (k < KMAX) ? v : 0.0f;         // v_cndmask, no exec divergence
    }
    f[e] = (_Float16)v;
  }
  return f;
}

// ---------------- kernel 0: zero the global accumulators ----------------
__global__ void osml_init_kernel(float* __restrict__ acc) {
  acc[0] = 0.0f;   // losses
  acc[1] = 0.0f;   // counts
}

// ---------------- kernel 1: G = labels*labels^T (diag 0), S = 0.5*(64 - H*H^T)
// one wave (32 threads) per 16x16 output tile; 256 tiles. 3 WMMAs per tile.
__global__ __launch_bounds__(32)
void osml_gram_wmma_kernel(const float* __restrict__ hash,
                           const float* __restrict__ labels,
                           float* __restrict__ G,
                           float* __restrict__ S) {
  const int tile = blockIdx.x;            // 0..255
  const int mt = tile >> 4, nt = tile & 15;
  const int l  = threadIdx.x & 31;        // 0..31, one full wave (EXEC all ones)
  const int hi = (l >> 4) & 1;            // provably 0 or 1
  const int ln = l & 15;

  const float* arow_h = hash   + (mt * 16 + ln) * NHASH;
  const float* brow_h = hash   + (nt * 16 + ln) * NHASH;
  const float* arow_l = labels + (mt * 16 + ln) * NCLS;
  const float* brow_l = labels + (nt * 16 + ln) * NCLS;

  // S tile: K = 64 -> two chained K=32 WMMAs (unconditional vector loads)
  v8f cS = {};
  {
    v16h a0 = frag_load_row<NHASH, true>(arow_h, hi, 0);
    v16h b0 = frag_load_row<NHASH, true>(brow_h, hi, 0);
    cS = __builtin_amdgcn_wmma_f32_16x16x32_f16(false, a0, false, b0,
                                                (short)0, cS, false, false);
    v16h a1 = frag_load_row<NHASH, true>(arow_h, hi, 32);
    v16h b1 = frag_load_row<NHASH, true>(brow_h, hi, 32);
    cS = __builtin_amdgcn_wmma_f32_16x16x32_f16(false, a1, false, b1,
                                                (short)0, cS, false, false);
  }
  // G tile: K = 14, zero-padded to 32 (0/1 labels -> exact in f16)
  v8f cG = {};
  {
    v16h a = frag_load_row<NCLS, false>(arow_l, hi, 0);
    v16h b = frag_load_row<NCLS, false>(brow_l, hi, 0);
    cG = __builtin_amdgcn_wmma_f32_16x16x32_f16(false, a, false, b,
                                                (short)0, cG, false, false);
  }

  // C/D layout: lane l, VGPR v -> (M = v + 8*(l>=16), N = l&15)
  const int n = nt * 16 + ln;
#pragma unroll
  for (int v = 0; v < 8; ++v) {
    const int m = mt * 16 + hi * 8 + v;
    S[m * NN + n] = 0.5f * ((float)NHASH - cS[v]);
    G[m * NN + n] = (m == n) ? 0.0f : cG[v];
  }
}

// ---------------- shared pair loop + block reduction ----------------
__device__ __forceinline__ void osml_pair_reduce(const float* g, const float* s,
                                                 const float* p2, int j,
                                                 float* __restrict__ acc) {
  const float gj = g[j], sj = s[j], pj = p2[j];
  float z = 0.0f, ls = 0.0f, cnt = 0.0f;
#pragma unroll 8
  for (int k = 0; k < NN; ++k) {
    const float gk = g[k];
    if (gj > gk) {                       // mask
      const float r = pj - p2[k];        // rij2 = 2^gj - 2^gk  (> 0 under mask)
      z += r;
      if (gj - gk <= 2.0f) {             // keep
        const float t = sj - s[k] + 5.0f;
        ls = fmaf(r, fmaxf(t, 0.0f), ls);
        cnt += 1.0f;                     // counts fit exactly in f32 (<= 2^24)
      }
    }
  }
  __shared__ float rz[NN], rl[NN], rc[NN];
  rz[j] = z; rl[j] = ls; rc[j] = cnt;
  __syncthreads();
  for (int off = NN / 2; off > 0; off >>= 1) {
    if (j < off) { rz[j] += rz[j + off]; rl[j] += rl[j + off]; rc[j] += rc[j + off]; }
    __syncthreads();
  }
  if (j == 0) {
    const float contrib = (rz[0] > 0.0f) ? (rl[0] / rz[0]) : 0.0f;
    atomicAdd(&acc[0], contrib);
    atomicAdd(&acc[1], rc[0]);
  }
}

// ---------------- kernel 2: one block per anchor, rows cached in LDS ----------------
__global__ __launch_bounds__(NN)
void osml_pair_kernel(const float* __restrict__ G, const float* __restrict__ S,
                      float* __restrict__ acc) {
  const int i = blockIdx.x;
  const int j = threadIdx.x;
  __shared__ float g[NN], s[NN], p2[NN];
  g[j]  = G[i * NN + j];
  s[j]  = S[i * NN + j];
  p2[j] = exp2f(g[j]);                   // g is an exact small integer
  __syncthreads();
  osml_pair_reduce(g, s, p2, j, acc);
}

// ---------------- fallback (tiny workspace): fused row compute, no G/S scratch ----------------
__global__ __launch_bounds__(NN)
void osml_fused_fallback_kernel(const float* __restrict__ hash,
                                const float* __restrict__ labels,
                                float* __restrict__ acc) {
  const int i = blockIdx.x;
  const int j = threadIdx.x;
  __shared__ float g[NN], s[NN], p2[NN];
  {
    const float* li = labels + i * NCLS;
    const float* lj = labels + j * NCLS;
    float d = 0.0f;
#pragma unroll
    for (int c = 0; c < NCLS; ++c) d = fmaf(li[c], lj[c], d);
    g[j] = (i == j) ? 0.0f : d;
    const float* hri = hash + i * NHASH;
    const float* hrj = hash + j * NHASH;
    float hd = 0.0f;
#pragma unroll
    for (int c = 0; c < NHASH; ++c) hd = fmaf(hri[c], hrj[c], hd);
    s[j]  = 0.5f * ((float)NHASH - hd);
    p2[j] = exp2f(g[j]);
  }
  __syncthreads();
  osml_pair_reduce(g, s, p2, j, acc);
}

// ---------------- kernel 3: finalize scalar ----------------
__global__ void osml_finalize_kernel(const float* __restrict__ acc,
                                     float* __restrict__ out) {
  const float c = acc[1];
  out[0] = (c > 0.0f) ? (acc[0] / c) : acc[0];
}

extern "C" void kernel_launch(void* const* d_in, const int* in_sizes, int n_in,
                              void* d_out, int out_size, void* d_ws, size_t ws_size,
                              hipStream_t stream) {
  const float* hash   = (const float*)d_in[0];   // 256 x 64  f32
  const float* labels = (const float*)d_in[1];   // 256 x 14  f32
  float* out = (float*)d_out;                    // scalar f32

  float* wsf = (float*)d_ws;
  float* acc = wsf;                              // [0]=losses, [1]=counts
  float* G   = wsf + 16;                         // 256x256 f32
  float* S   = G + NN * NN;                      // 256x256 f32
  const size_t needed = (size_t)(16 + 2 * NN * NN) * sizeof(float);

  hipLaunchKernelGGL(osml_init_kernel, dim3(1), dim3(1), 0, stream, acc);
  if (ws_size >= needed) {
    hipLaunchKernelGGL(osml_gram_wmma_kernel, dim3(256), dim3(32), 0, stream,
                       hash, labels, G, S);
    hipLaunchKernelGGL(osml_pair_kernel, dim3(NN), dim3(NN), 0, stream, G, S, acc);
  } else {
    hipLaunchKernelGGL(osml_fused_fallback_kernel, dim3(NN), dim3(NN), 0, stream,
                       hash, labels, acc);
  }
  hipLaunchKernelGGL(osml_finalize_kernel, dim3(1), dim3(1), 0, stream, acc, out);
}
